// MLPFewParams_68762426409368
// MI455X (gfx1250) — matrix operand order repaired and checked by reading, not compiled, not run on hardware
//
#include <hip/hip_runtime.h>
#include <hip/hip_bf16.h>

// MI455X / gfx1250, wave32. B=256, T=128, C=768, R=64.
// HBM-bound (~1.5 GB traffic -> ~63us floor @ 23.3 TB/s).
// Weights pre-converted to bf16 in d_ws (one-time); GEMMs via
// v_wmma_f32_16x16x32_bf16; front/back/x/bias streamed non-temporally.

typedef __attribute__((ext_vector_type(16))) __bf16 v16bf;
typedef __attribute__((ext_vector_type(8)))  float  v8f;
typedef __attribute__((ext_vector_type(4)))  float  fx4;

#define C_DIM 768
#define R_DIM 64
#define TOK_PER_BLK 16
#define W_ELEMS (R_DIM * C_DIM)          // 49152 per weight matrix

static __device__ __forceinline__ unsigned short f2bf(float f) {
    unsigned int u = __float_as_uint(f);
    u += 0x7FFFu + ((u >> 16) & 1u);     // round-to-nearest-even
    return (unsigned short)(u >> 16);
}
static __device__ __forceinline__ unsigned int pk2bf(float lo, float hi) {
    return (unsigned int)f2bf(lo) | ((unsigned int)f2bf(hi) << 16);
}

static __device__ __forceinline__ v8f wmma_bf16(v16bf a, v16bf b, v8f c) {
    return __builtin_amdgcn_wmma_f32_16x16x32_bf16(
        false, a, false, b, (short)0, c, false, false);
}

// A-fragment (16x32 bf16, M x K) from LDS row-major bf16 matrix.
// Lane l<16: row=l,     K = ks*32 + 0..15
// Lane l>=16: row=l-16, K = ks*32 + 16..31
static __device__ __forceinline__ v16bf load_A_lds(const unsigned short* base,
                                                   int rowStride, int ks, int lane) {
    const int row = lane & 15;
    const int o1  = ks * 32 + ((lane & 16) ? 16 : 0);
    const unsigned short* p = base + row * rowStride + o1;
    union { v16bf v; uint4 q[2]; } r;
    r.q[0] = *(const uint4*)(p);
    r.q[1] = *(const uint4*)(p + 8);
    return r.v;
}

// B-fragment (32x16 bf16, K x N) from pre-converted bf16 weight, (N x K) row-major.
// Lane n<16: col=n, K = ks*32 + 0..15 ; lane n>=16: col=n-16, K = ks*32 + 16..31
static __device__ __forceinline__ v16bf load_B_bf(const unsigned short* w, int Kstride,
                                                  int nBase, int ks, int lane) {
    const int n  = nBase + (lane & 15);
    const int kb = ks * 32 + ((lane & 16) ? 16 : 0);
    const unsigned short* p = w + (size_t)n * Kstride + kb;
    union { v16bf v; uint4 q[2]; } r;
    r.q[0] = *(const uint4*)(p);
    r.q[1] = *(const uint4*)(p + 8);
    return r.v;
}

static __device__ __forceinline__ float gelu_tanh(float x) {
    float x3 = x * x * x;
    float t  = tanhf(0.7978845608028654f * (x + 0.044715f * x3));
    return 0.5f * x * (1.0f + t);
}

// ---- One-time weight conversion: {fU, fV, bU, bV} f32 -> bf16 in ws ----
__global__ __launch_bounds__(256)
void convert_weights_kernel(const float* __restrict__ fU, const float* __restrict__ fV,
                            const float* __restrict__ bU, const float* __restrict__ bV,
                            unsigned short* __restrict__ ws) {
    const int idx = blockIdx.x * 256 + threadIdx.x;     // one float4 per thread
    const int e   = idx * 4;                            // element index, 0..196607
    const int a   = e / W_ELEMS;
    const int off = e % W_ELEMS;
    const float* srcs[4] = {fU, fV, bU, bV};
    fx4 f = __builtin_nontemporal_load((const fx4*)(srcs[a] + off));
    uint2 pk;
    pk.x = pk2bf(f.x, f.y);
    pk.y = pk2bf(f.z, f.w);
    ((uint2*)ws)[idx] = pk;
}

__global__ __launch_bounds__(256)
void mlp_lowrank_kernel(const float* __restrict__ x,
                        const float* __restrict__ front,
                        const float* __restrict__ back,
                        const float* __restrict__ hiddenBias,
                        const unsigned short* __restrict__ wbf,  // fU|fV|bU|bV bf16
                        float* __restrict__ out) {
    __shared__ unsigned short Xbf[TOK_PER_BLK * C_DIM];   // 24 KB
    __shared__ unsigned short Hbf[TOK_PER_BLK * C_DIM];   // 24 KB
    __shared__ float          h1 [TOK_PER_BLK * R_DIM];   // 4 KB
    __shared__ float          g1 [TOK_PER_BLK * R_DIM];   // 4 KB
    __shared__ unsigned short h2bf[TOK_PER_BLK * R_DIM];  // 2 KB
    __shared__ unsigned short g2bf[TOK_PER_BLK * R_DIM];  // 2 KB

    const unsigned short* fUb = wbf;
    const unsigned short* fVb = wbf + 1 * W_ELEMS;
    const unsigned short* bUb = wbf + 2 * W_ELEMS;
    const unsigned short* bVb = wbf + 3 * W_ELEMS;

    const int tid  = threadIdx.x;
    const int lane = tid & 31;
    const int wid  = tid >> 5;
    const size_t tok0 = (size_t)blockIdx.x * TOK_PER_BLK;

    // ---- Stage 0: stage x tile -> LDS bf16; zero reduction buffers ----
    {
        const fx4* xsrc = (const fx4*)(x + tok0 * C_DIM);
        #pragma unroll
        for (int i = 0; i < 12; ++i) {               // 16*768/4 = 3072 float4
            int e4 = tid + i * 256;
            fx4 f = __builtin_nontemporal_load(xsrc + e4);
            int e = e4 * 4;
            Xbf[e + 0] = f2bf(f.x); Xbf[e + 1] = f2bf(f.y);
            Xbf[e + 2] = f2bf(f.z); Xbf[e + 3] = f2bf(f.w);
        }
        #pragma unroll
        for (int i = 0; i < 4; ++i) {
            h1[tid + i * 256] = 0.0f;
            g1[tid + i * 256] = 0.0f;
        }
        // Prefetch upcoming 'front' tile toward L2 (global_prefetch_b8).
        const float* fpre = front + tok0 * (R_DIM * R_DIM);
        #pragma unroll
        for (int i = 0; i < 4; ++i)
            __builtin_prefetch(fpre + (size_t)(tid + i * 256) * 64, 0, 1);
    }
    __syncthreads();

    // ---- Stage 1: H1 = X @ fU^T   (M=16, N=64, K=768) ----
    {
        const int nt = wid & 3;        // N-tile 0..3
        const int kh = wid >> 2;       // K-half 0..1 (12 ksteps each)
        v8f c = {};
        for (int ks = kh * 12; ks < kh * 12 + 12; ++ks) {
            v16bf a = load_A_lds(Xbf, C_DIM, ks, lane);
            v16bf b = load_B_bf(fUb, C_DIM, nt * 16, ks, lane);
            c = wmma_bf16(a, b, c);
        }
        const int kcol = nt * 16 + (lane & 15);
        const int mb   = (lane >> 4) * 8;
        #pragma unroll
        for (int r = 0; r < 8; ++r)
            atomicAdd(&h1[(mb + r) * R_DIM + kcol], c[r]);
    }
    __syncthreads();

    // ---- Stage 2: per-token h2 = cf @ h1 (streams 'front', non-temporal) ----
    {
        const int tok = tid >> 4;
        const int kr  = tid & 15;
        const float* cf = front + (tok0 + tok) * (R_DIM * R_DIM);
        float acc[4] = {0.f, 0.f, 0.f, 0.f};
        for (int j = 0; j < R_DIM; j += 4) {
            fx4 hv = *(const fx4*)&h1[tok * R_DIM + j];
            #pragma unroll
            for (int q = 0; q < 4; ++q) {
                fx4 m = __builtin_nontemporal_load(
                    (const fx4*)(cf + (kr + 16 * q) * R_DIM + j));
                acc[q] += m.x * hv.x + m.y * hv.y + m.z * hv.z + m.w * hv.w;
            }
        }
        #pragma unroll
        for (int q = 0; q < 4; ++q)
            h2bf[tok * R_DIM + kr + 16 * q] = f2bf(acc[q]);
    }
    __syncthreads();

    // ---- Stage 3: H = gelu(H2 @ fV^T + bias) -> LDS bf16 (M=16, N=768, K=64) ----
    {
        v16bf a0 = load_A_lds(h2bf, R_DIM, 0, lane);
        v16bf a1 = load_A_lds(h2bf, R_DIM, 1, lane);
        for (int i = 0; i < 6; ++i) {
            const int nt = wid + i * 8;            // 0..47
            v8f c = {};
            c = wmma_bf16(a0, load_B_bf(fVb, R_DIM, nt * 16, 0, lane), c);
            c = wmma_bf16(a1, load_B_bf(fVb, R_DIM, nt * 16, 1, lane), c);
            const int ncol = nt * 16 + (lane & 15);
            const int mb   = (lane >> 4) * 8;
            #pragma unroll
            for (int r = 0; r < 8; ++r) {
                size_t gi = (tok0 + mb + r) * C_DIM + ncol;
                float bia = __builtin_nontemporal_load(hiddenBias + gi);
                float v = gelu_tanh(c[r] + bia);
                Hbf[(mb + r) * C_DIM + ncol] = f2bf(v);
            }
        }
    }
    __syncthreads();

    // ---- Stage 4: G1 = H @ bU^T  (M=16, N=64, K=768) ----
    {
        const int nt = wid & 3;
        const int kh = wid >> 2;
        v8f c = {};
        for (int ks = kh * 12; ks < kh * 12 + 12; ++ks) {
            v16bf a = load_A_lds(Hbf, C_DIM, ks, lane);
            v16bf b = load_B_bf(bUb, C_DIM, nt * 16, ks, lane);
            c = wmma_bf16(a, b, c);
        }
        const int kcol = nt * 16 + (lane & 15);
        const int mb   = (lane >> 4) * 8;
        #pragma unroll
        for (int r = 0; r < 8; ++r)
            atomicAdd(&g1[(mb + r) * R_DIM + kcol], c[r]);
    }
    __syncthreads();

    // ---- Stage 5: per-token g2 = cb @ g1 (streams 'back', non-temporal) ----
    {
        const int tok = tid >> 4;
        const int kr  = tid & 15;
        const float* cb = back + (tok0 + tok) * (R_DIM * R_DIM);
        float acc[4] = {0.f, 0.f, 0.f, 0.f};
        for (int j = 0; j < R_DIM; j += 4) {
            fx4 gv = *(const fx4*)&g1[tok * R_DIM + j];
            #pragma unroll
            for (int q = 0; q < 4; ++q) {
                fx4 m = __builtin_nontemporal_load(
                    (const fx4*)(cb + (kr + 16 * q) * R_DIM + j));
                acc[q] += m.x * gv.x + m.y * gv.y + m.z * gv.z + m.w * gv.w;
            }
        }
        #pragma unroll
        for (int q = 0; q < 4; ++q)
            g2bf[tok * R_DIM + kr + 16 * q] = f2bf(acc[q]);
    }
    __syncthreads();

    // ---- Stage 6: out = G2 @ bV^T  (M=16, N=768, K=64) -> global f32 (NT) ----
    {
        v16bf a0 = load_A_lds(g2bf, R_DIM, 0, lane);
        v16bf a1 = load_A_lds(g2bf, R_DIM, 1, lane);
        for (int i = 0; i < 6; ++i) {
            const int nt = wid + i * 8;
            v8f c = {};
            c = wmma_bf16(a0, load_B_bf(bVb, R_DIM, nt * 16, 0, lane), c);
            c = wmma_bf16(a1, load_B_bf(bVb, R_DIM, nt * 16, 1, lane), c);
            const int ncol = nt * 16 + (lane & 15);
            const int mb   = (lane >> 4) * 8;
            #pragma unroll
            for (int r = 0; r < 8; ++r)
                __builtin_nontemporal_store(c[r],
                    out + (tok0 + mb + r) * C_DIM + ncol);
        }
    }
}

extern "C" void kernel_launch(void* const* d_in, const int* in_sizes, int n_in,
                              void* d_out, int out_size, void* d_ws, size_t ws_size,
                              hipStream_t stream) {
    const float* x          = (const float*)d_in[0];
    const float* front      = (const float*)d_in[1];
    const float* back       = (const float*)d_in[2];
    const float* hiddenBias = (const float*)d_in[3];
    const float* fU         = (const float*)d_in[4];
    const float* fV         = (const float*)d_in[5];
    const float* bU         = (const float*)d_in[6];
    const float* bV         = (const float*)d_in[7];
    float* out = (float*)d_out;
    unsigned short* wbf = (unsigned short*)d_ws;     // 4 * 49152 bf16 = 384 KB

    // One-time (per launch) weight f32->bf16 conversion into workspace.
    // 4*49152 elements / 4 per thread / 256 threads = 192 blocks.
    convert_weights_kernel<<<192, 256, 0, stream>>>(fU, fV, bU, bV, wbf);

    const int tokens = in_sizes[0] / C_DIM;          // B*T = 32768
    const int grid   = tokens / TOK_PER_BLK;         // 2048 blocks
    mlp_lowrank_kernel<<<grid, 256, 0, stream>>>(
        x, front, back, hiddenBias, wbf, out);
}
